// Eff_FactorAtt_ConvRelPosEnc_21096879358336
// MI455X (gfx1250) — matrix operand order repaired and checked by reading, not compiled, not run on hardware
//
#include <hip/hip_runtime.h>
#include <hip/hip_bf16.h>

typedef __bf16 bf16;
typedef __attribute__((ext_vector_type(8)))  __bf16 v8bf;
typedef __attribute__((ext_vector_type(16))) __bf16 v16bf;
typedef __attribute__((ext_vector_type(8)))  float  v8f;

constexpr int kB  = 16;
constexpr int kN  = 4096;
constexpr int kC  = 256;
constexpr int kH  = 8;
constexpr int kHD = 32;
constexpr int kM  = kB * kN;      // 65536 rows
constexpr int kQKV = 3 * kC;      // 768
constexpr int kLdsStride = 40;    // 80B row stride: fragment rows hit banks
                                  // 0,4,...,60 -> conflict-free b128 reads

// Build a 16-element bf16 WMMA fragment from two contiguous 8-element (16B) loads.
__device__ __forceinline__ v16bf frag_from(const bf16* p0, const bf16* p1) {
  v8bf lo = *(const v8bf*)p0;
  v8bf hi = *(const v8bf*)p1;
  v16bf r;
#pragma unroll
  for (int i = 0; i < 8; ++i) { r[i] = lo[i]; r[i + 8] = hi[i]; }
  return r;
}

__device__ __forceinline__ v8f wmma_bf16(v16bf a, v16bf b, v8f c) {
  return __builtin_amdgcn_wmma_f32_16x16x32_bf16(false, a, false, b, (short)0, c,
                                                 false, false);
}

// ---------------------------------------------------------------------------
// Kernel 1: qkv = x @ W_qkv^T      (M=65536, K=256, N=768), fp32 out
// Block: 256 thr (8 waves). Tile 128(M) x 64(N), K-step 32; 32x32 per wave
// (4 WMMAs / wave / K-step against 8 ds_load_b128 -> 2:1 LDS:WMMA).
// ---------------------------------------------------------------------------
__global__ __launch_bounds__(256) void k_gemm_qkv(const float* __restrict__ X,
                                                  const float* __restrict__ W,
                                                  float* __restrict__ QKV) {
  __shared__ __align__(16) bf16 As[128][kLdsStride];
  __shared__ __align__(16) bf16 Bs[64][kLdsStride];
  const int ntn = kQKV / 64;                 // 12
  const int tile_m = blockIdx.x / ntn;
  const int tile_n = blockIdx.x % ntn;
  const int tid  = threadIdx.x;
  const int lane = tid & 31;
  const int wave = tid >> 5;
  const int m_off = (wave & 3) * 32;         // 4 waves tile M
  const int n_off = (wave >> 2) * 32;        // 2 waves tile N
  const int ar = tid >> 1;                   // A staging: row 0..127
  const int ac = (tid & 1) * 16;             // A staging: col 0 / 16
  const int br = tid >> 2;                   // B staging: row 0..63
  const int bc = (tid & 3) * 8;              // B staging: col 0,8,16,24
  const int rl  = lane & 15;
  const int kb  = (lane < 16) ? 0 : 8;       // A-fragment K base
  const int kbb = (lane < 16) ? 0 : 8;       // B-fragment K base
  v8f acc00 = {}, acc01 = {}, acc10 = {}, acc11 = {};

  for (int k0 = 0; k0 < kC; k0 += 32) {
    const float* pa = X + (size_t)(tile_m * 128 + ar) * kC + k0 + ac;
    const float* pb = W + (size_t)(tile_n * 64 + br) * kC + k0 + bc;
    if (k0 + 32 < kC) {
      __builtin_prefetch(pa + 32, 0, 3);     // global_prefetch_b8, near scope
      __builtin_prefetch(pb + 32, 0, 3);
    }
#pragma unroll
    for (int i = 0; i < 16; ++i) As[ar][ac + i] = (bf16)pa[i];
#pragma unroll
    for (int i = 0; i < 8; ++i)  Bs[br][bc + i] = (bf16)pb[i];
    __syncthreads();

    v16bf a0 = frag_from(&As[m_off + rl][kb],       &As[m_off + rl][kb + 16]);
    v16bf a1 = frag_from(&As[m_off + 16 + rl][kb],  &As[m_off + 16 + rl][kb + 16]);
    v16bf b0 = frag_from(&Bs[n_off + rl][kbb],      &Bs[n_off + rl][kbb + 16]);
    v16bf b1 = frag_from(&Bs[n_off + 16 + rl][kbb], &Bs[n_off + 16 + rl][kbb + 16]);
    acc00 = wmma_bf16(a0, b0, acc00);
    acc01 = wmma_bf16(a0, b1, acc01);
    acc10 = wmma_bf16(a1, b0, acc10);
    acc11 = wmma_bf16(a1, b1, acc11);
    __syncthreads();
  }
  // D layout: VGPR j -> (M=j, N=lane) for lane<16 ; (M=j+8, N=lane-16) else
  const int gm0 = tile_m * 128 + m_off + ((lane < 16) ? 0 : 8);
  const int gn  = tile_n * 64 + n_off + rl;
#pragma unroll
  for (int j = 0; j < 8; ++j) {
    QKV[(size_t)(gm0 + j) * kQKV + gn]           = acc00[j];
    QKV[(size_t)(gm0 + j) * kQKV + gn + 16]      = acc01[j];
    QKV[(size_t)(gm0 + 16 + j) * kQKV + gn]      = acc10[j];
    QKV[(size_t)(gm0 + 16 + j) * kQKV + gn + 16] = acc11[j];
  }
}

// ---------------------------------------------------------------------------
// Kernel 2: k softmax over tokens (axis N) + v transpose.
// One block per (b, c); writes bf16 kT/vT in [B, h, hd, N] (channel-major).
// Adjacent blocks (c, c+1, ...) read adjacent columns -> same 128B lines,
// absorbed by the 192MB L2 (qkv is L2-resident).
// ---------------------------------------------------------------------------
__global__ __launch_bounds__(256) void k_ksm_vt(const float* __restrict__ QKV,
                                                bf16* __restrict__ kT,
                                                bf16* __restrict__ vT) {
  __shared__ float red[256];
  const int b = blockIdx.x >> 8;             // /256
  const int c = blockIdx.x & 255;
  const int t = threadIdx.x;
  const float* base = QKV + (size_t)b * kN * kQKV;
  float vals[16];
  float mx = -1e30f;
#pragma unroll
  for (int i = 0; i < 16; ++i) {
    const int n = i * 256 + t;
    const float v = base[(size_t)n * kQKV + kC + c];
    vals[i] = v;
    mx = fmaxf(mx, v);
  }
  red[t] = mx; __syncthreads();
  for (int s = 128; s > 0; s >>= 1) {
    if (t < s) red[t] = fmaxf(red[t], red[t + s]);
    __syncthreads();
  }
  mx = red[0]; __syncthreads();
  float sum = 0.f;
#pragma unroll
  for (int i = 0; i < 16; ++i) { vals[i] = __expf(vals[i] - mx); sum += vals[i]; }
  red[t] = sum; __syncthreads();
  for (int s = 128; s > 0; s >>= 1) {
    if (t < s) red[t] += red[t + s];
    __syncthreads();
  }
  const float inv = 1.f / red[0];
  bf16* kout = kT + ((size_t)b * kC + c) * kN;
  bf16* vout = vT + ((size_t)b * kC + c) * kN;
#pragma unroll
  for (int i = 0; i < 16; ++i) {
    const int n = i * 256 + t;
    kout[n] = (bf16)(vals[i] * inv);
    vout[n] = (bf16)base[(size_t)n * kQKV + 2 * kC + c];
  }
}

// ---------------------------------------------------------------------------
// Kernel 3: q softmax over hd=32 (one wave32 per (b,n,h), lane = channel).
// Writes bf16 qs in [B, h, N, hd] (token-major rows of 32).
// ---------------------------------------------------------------------------
__global__ __launch_bounds__(256) void k_qsm(const float* __restrict__ QKV,
                                             bf16* __restrict__ qs) {
  const int w    = blockIdx.x * 8 + (threadIdx.x >> 5);
  const int lane = threadIdx.x & 31;
  const int b   = w >> 15;                   // /(4096*8)
  const int rem = w & 32767;
  const int n   = rem >> 3;
  const int h   = rem & 7;
  const float v = QKV[(size_t)(b * kN + n) * kQKV + h * kHD + lane];
  float m = v;
#pragma unroll
  for (int off = 16; off > 0; off >>= 1) m = fmaxf(m, __shfl_xor(m, off, 32));
  const float e = __expf(v - m);
  float s = e;
#pragma unroll
  for (int off = 16; off > 0; off >>= 1) s += __shfl_xor(s, off, 32);
  qs[((size_t)(b * kH + h) * kN + n) * kHD + lane] = (bf16)(e / s);
}

// ---------------------------------------------------------------------------
// Kernel 4: context[kd,vd] = sum_n k_sm[kd,n] * v[vd,n]   per (b,h).
// Block of 8 waves per (b,h): each wave reduces 512 tokens with 4 WMMA tiles,
// then fp32 partials are reduced across waves through LDS.
// Stores bf16 transposed: ctxT[vd][kd]  (so attended-B columns are contiguous).
// ---------------------------------------------------------------------------
__global__ __launch_bounds__(256) void k_context(const bf16* __restrict__ kT,
                                                 const bf16* __restrict__ vT,
                                                 bf16* __restrict__ ctxT) {
  __shared__ float part[8][kHD * kHD];       // 32 KB fp32 partial tiles
  const int bh   = blockIdx.x;               // 0..127
  const int tid  = threadIdx.x;
  const int lane = tid & 31;
  const int wave = tid >> 5;                 // 0..7 : slice of N
  const int rl   = lane & 15;
  const int kb   = (lane < 16) ? 0 : 8;
  const int kbb  = (lane < 16) ? 0 : 8;
  const bf16* kp = kT + (size_t)bh * kHD * kN;
  const bf16* vp = vT + (size_t)bh * kHD * kN;
  v8f acc[4] = {{}, {}, {}, {}};

  const int nbeg = wave * (kN / 8);          // 512 tokens per wave
  for (int it = 0; it < (kN / 8); it += 32) {
    const int n0 = nbeg + it;
    v16bf a0 = frag_from(kp + (size_t)rl * kN + n0 + kb,
                         kp + (size_t)rl * kN + n0 + kb + 16);
    v16bf a1 = frag_from(kp + (size_t)(16 + rl) * kN + n0 + kb,
                         kp + (size_t)(16 + rl) * kN + n0 + kb + 16);
    v16bf b0 = frag_from(vp + (size_t)rl * kN + n0 + kbb,
                         vp + (size_t)rl * kN + n0 + kbb + 16);
    v16bf b1 = frag_from(vp + (size_t)(16 + rl) * kN + n0 + kbb,
                         vp + (size_t)(16 + rl) * kN + n0 + kbb + 16);
    acc[0] = wmma_bf16(a0, b0, acc[0]);
    acc[1] = wmma_bf16(a0, b1, acc[1]);
    acc[2] = wmma_bf16(a1, b0, acc[2]);
    acc[3] = wmma_bf16(a1, b1, acc[3]);
  }
  // dump per-wave partial tiles to LDS in [kd][vd] order
#pragma unroll
  for (int i = 0; i < 2; ++i)
#pragma unroll
    for (int j = 0; j < 2; ++j)
#pragma unroll
      for (int jj = 0; jj < 8; ++jj) {
        const int kd = i * 16 + jj + ((lane < 16) ? 0 : 8);
        const int vd = j * 16 + rl;
        part[wave][kd * kHD + vd] = acc[i * 2 + j][jj];
      }
  __syncthreads();
  // reduce 8 partials; store transposed ctxT[vd][kd]
  bf16* cp = ctxT + (size_t)bh * kHD * kHD;
#pragma unroll
  for (int i = 0; i < 4; ++i) {
    const int l  = tid + 256 * i;            // output index = vd*32 + kd
    const int vd = l >> 5;
    const int kd = l & 31;
    float s = 0.f;
#pragma unroll
    for (int w8 = 0; w8 < 8; ++w8) s += part[w8][kd * kHD + vd];
    cp[l] = (bf16)s;
  }
}

// ---------------------------------------------------------------------------
// Kernel 5: attended[n,vd] = sum_kd q_sm[n,kd] * context[kd,vd]  (K = 32, one
// WMMA per 16x16 tile). One wave per (b,h, 16-token tile). Output written in
// [B*N, C] bf16 with channel c = h*32+vd -> directly the "agg" conv input.
// ---------------------------------------------------------------------------
__global__ __launch_bounds__(256) void k_attended(const bf16* __restrict__ qs,
                                                  const bf16* __restrict__ ctxT,
                                                  bf16* __restrict__ agg) {
  const int w    = blockIdx.x * 8 + (threadIdx.x >> 5);   // 0..32767
  const int lane = threadIdx.x & 31;
  const int bh = w >> 8;
  const int mt = w & 255;
  const int n0 = mt * 16;
  const int rl  = lane & 15;
  const int kb  = (lane < 16) ? 0 : 8;
  const int kbb = (lane < 16) ? 0 : 8;
  const bf16* qp = qs + ((size_t)bh * kN + n0) * kHD;
  const bf16* cp = ctxT + (size_t)bh * kHD * kHD;

  v16bf a  = frag_from(qp + (size_t)rl * kHD + kb,  qp + (size_t)rl * kHD + kb + 16);
  v16bf b0 = frag_from(cp + (size_t)rl * kHD + kbb, cp + (size_t)rl * kHD + kbb + 16);
  v16bf b1 = frag_from(cp + (size_t)(16 + rl) * kHD + kbb,
                       cp + (size_t)(16 + rl) * kHD + kbb + 16);
  v8f acc0 = {}, acc1 = {};
  acc0 = wmma_bf16(a, b0, acc0);
  acc1 = wmma_bf16(a, b1, acc1);

  const int b = bh >> 3, h = bh & 7;
#pragma unroll
  for (int jj = 0; jj < 8; ++jj) {
    const int tl = jj + ((lane < 16) ? 0 : 8);
    const size_t row = (size_t)(b * kN + n0 + tl) * kC + h * kHD;
    agg[row + rl]      = (bf16)acc0[jj];
    agg[row + 16 + rl] = (bf16)acc1[jj];
  }
}

// ---------------------------------------------------------------------------
// Kernel 6: out = agg @ W_conv^T + b_conv   (M=65536, K=256, N=256), fp32 out.
// Same 128x64 tiling as kernel 1; A is already bf16.
// ---------------------------------------------------------------------------
__global__ __launch_bounds__(256) void k_gemm_out(const bf16* __restrict__ AG,
                                                  const float* __restrict__ Wc,
                                                  const float* __restrict__ bias,
                                                  float* __restrict__ OUT) {
  __shared__ __align__(16) bf16 As[128][kLdsStride];
  __shared__ __align__(16) bf16 Bs[64][kLdsStride];
  const int ntn = kC / 64;                   // 4
  const int tile_m = blockIdx.x / ntn;
  const int tile_n = blockIdx.x % ntn;
  const int tid  = threadIdx.x;
  const int lane = tid & 31;
  const int wave = tid >> 5;
  const int m_off = (wave & 3) * 32;
  const int n_off = (wave >> 2) * 32;
  const int ar = tid >> 1;
  const int ac = (tid & 1) * 16;
  const int br = tid >> 2;
  const int bc = (tid & 3) * 8;
  const int rl  = lane & 15;
  const int kb  = (lane < 16) ? 0 : 8;
  const int kbb = (lane < 16) ? 0 : 8;
  v8f acc00 = {}, acc01 = {}, acc10 = {}, acc11 = {};

  for (int k0 = 0; k0 < kC; k0 += 32) {
    const bf16*  pa = AG + (size_t)(tile_m * 128 + ar) * kC + k0 + ac;
    const float* pb = Wc + (size_t)(tile_n * 64 + br) * kC + k0 + bc;
    if (k0 + 32 < kC) {
      __builtin_prefetch(pa + 32, 0, 3);
      __builtin_prefetch(pb + 32, 0, 3);
    }
    v8bf va0 = *(const v8bf*)pa;
    v8bf va1 = *(const v8bf*)(pa + 8);
#pragma unroll
    for (int i = 0; i < 8; ++i) { As[ar][ac + i] = va0[i]; As[ar][ac + 8 + i] = va1[i]; }
#pragma unroll
    for (int i = 0; i < 8; ++i) Bs[br][bc + i] = (bf16)pb[i];
    __syncthreads();

    v16bf a0 = frag_from(&As[m_off + rl][kb],       &As[m_off + rl][kb + 16]);
    v16bf a1 = frag_from(&As[m_off + 16 + rl][kb],  &As[m_off + 16 + rl][kb + 16]);
    v16bf b0 = frag_from(&Bs[n_off + rl][kbb],      &Bs[n_off + rl][kbb + 16]);
    v16bf b1 = frag_from(&Bs[n_off + 16 + rl][kbb], &Bs[n_off + 16 + rl][kbb + 16]);
    acc00 = wmma_bf16(a0, b0, acc00);
    acc01 = wmma_bf16(a0, b1, acc01);
    acc10 = wmma_bf16(a1, b0, acc10);
    acc11 = wmma_bf16(a1, b1, acc11);
    __syncthreads();
  }
  const int gm0 = tile_m * 128 + m_off + ((lane < 16) ? 0 : 8);
  const int gn  = tile_n * 64 + n_off + rl;
  const float bi0 = bias[gn];
  const float bi1 = bias[gn + 16];
#pragma unroll
  for (int j = 0; j < 8; ++j) {
    OUT[(size_t)(gm0 + j) * kC + gn]           = acc00[j] + bi0;
    OUT[(size_t)(gm0 + j) * kC + gn + 16]      = acc01[j] + bi1;
    OUT[(size_t)(gm0 + 16 + j) * kC + gn]      = acc10[j] + bi0;
    OUT[(size_t)(gm0 + 16 + j) * kC + gn + 16] = acc11[j] + bi1;
  }
}

// ---------------------------------------------------------------------------
extern "C" void kernel_launch(void* const* d_in, const int* in_sizes, int n_in,
                              void* d_out, int out_size, void* d_ws, size_t ws_size,
                              hipStream_t stream) {
  const float* x  = (const float*)d_in[0];   // [16, 4096, 256]
  const float* Wq = (const float*)d_in[1];   // [768, 256]
  const float* Wc = (const float*)d_in[2];   // [256, 256]
  const float* bc = (const float*)d_in[3];   // [256]
  (void)in_sizes; (void)n_in; (void)out_size; (void)ws_size;

  char* ws = (char*)d_ws;
  size_t o = 0;
  float* qkv = (float*)(ws + o); o += (size_t)kM * kQKV * sizeof(float);      // 192 MiB
  bf16* kT   = (bf16*)(ws + o);  o += (size_t)kB * kC * kN * sizeof(bf16);    // 32 MiB
  bf16* vT   = (bf16*)(ws + o);  o += (size_t)kB * kC * kN * sizeof(bf16);    // 32 MiB
  bf16* qs   = (bf16*)(ws + o);  o += (size_t)kB * kC * kN * sizeof(bf16);    // 32 MiB
  bf16* ctxT = (bf16*)(ws + o);  o += (size_t)kB * kH * kHD * kHD * sizeof(bf16);
  bf16* agg  = (bf16*)(ws + o);  o += (size_t)kM * kC * sizeof(bf16);         // 32 MiB
  float* out = (float*)d_out;                                                 // [16,4096,256]

  k_gemm_qkv<<<dim3((kM / 128) * (kQKV / 64)), dim3(256), 0, stream>>>(x, Wq, qkv);
  k_ksm_vt  <<<dim3(kB * kC),                  dim3(256), 0, stream>>>(qkv, kT, vT);
  k_qsm     <<<dim3(kM * kH / 8),              dim3(256), 0, stream>>>(qkv, qs);
  k_context <<<dim3(kB * kH),                  dim3(256), 0, stream>>>(kT, vT, ctxT);
  k_attended<<<dim3(kM / 16 * kH / 8),         dim3(256), 0, stream>>>(qs, ctxT, agg);
  k_gemm_out<<<dim3((kM / 128) * (kC / 64)),   dim3(256), 0, stream>>>(agg, Wc, bc, out);
}
